// TmsFastAutoencoder_38276748542233
// MI455X (gfx1250) — compile-verified, matrix operand order checked by reading, not run on hardware
//
#include <hip/hip_runtime.h>

// ---------------------------------------------------------------------------
// TopK sparse autoencoder forward, MI455X (gfx1250, wave32, WMMA bf16)
//   B=4096, D=768, F=65536, k=64, auxk=256
// Encode GEMM is compute-bound (AI ~1900 FLOP/B) -> v_wmma_f32_16x16x32_bf16.
// Register-blocked: each wave computes a 32x16 output tile (2 row-tiles) so
// every B fragment fetched from L2/HBM feeds 2 WMMAs; K-loop is software-
// pipelined. A-strip is fetched with global_load_async_to_lds_b128 (ASYNCcnt)
// so the copy overlaps top-k list initialization with no VGPR staging.
// ---------------------------------------------------------------------------

#define BDIM 4096
#define DDIM 768
#define FDIM 65536
#define KTOP 64
#define AUXK 256
#define TM   32          // batch rows per workgroup (encode)
#define NF   128         // F columns per chunk (encode)
#define SSTRIDE 132      // staging row stride (pad vs bank conflicts)

typedef __attribute__((ext_vector_type(16))) __bf16 v16bf;
typedef __attribute__((ext_vector_type(8)))  __bf16 v8bf;
typedef __attribute__((ext_vector_type(8)))  float  v8f;

// ---------------------------------------------------------------- prep_x ----
__global__ __launch_bounds__(256) void prep_x_kernel(
    const float* __restrict__ x, const float* __restrict__ pre_bias,
    unsigned short* __restrict__ xd_out)
{
    __bf16* xd = reinterpret_cast<__bf16*>(xd_out);
    size_t i = (size_t)blockIdx.x * 256 + threadIdx.x;
    if (i < (size_t)BDIM * DDIM) {
        int dcol = (int)(i % DDIM);
        xd[i] = (__bf16)(x[i] - pre_bias[dcol]);
    }
}

// -------------------------------------------------------- transpose_wenc ----
// W_enc [D,F] f32 row-major  ->  W_encT [F,D] bf16 row-major (64x64 LDS tiles)
__global__ __launch_bounds__(256) void transpose_wenc_kernel(
    const float* __restrict__ W, unsigned short* __restrict__ WT_out)
{
    __bf16* WT = reinterpret_cast<__bf16*>(WT_out);
    __shared__ __bf16 tile[64][72];
    const int f0 = blockIdx.x * 64;
    const int d0 = blockIdx.y * 64;
    #pragma unroll
    for (int i = 0; i < 16; ++i) {
        int idx = threadIdx.x + i * 256;
        int r = idx >> 6, c = idx & 63;              // r: d-local, c: f-local
        tile[r][c] = (__bf16)W[(size_t)(d0 + r) * FDIM + (f0 + c)];
    }
    __syncthreads();
    #pragma unroll
    for (int i = 0; i < 16; ++i) {
        int idx = threadIdx.x + i * 256;
        int r = idx >> 6, c = idx & 63;              // r: f-local, c: d-local
        WT[(size_t)(f0 + r) * DDIM + (d0 + c)] = tile[c][r];
    }
}

// ------------------------------------------------------------ encode+topk ---
__global__ __launch_bounds__(256, 1) void encode_topk_kernel(
    const unsigned short* __restrict__ xd_in,     // [B,D] bf16
    const unsigned short* __restrict__ wt_in,     // [F,D] bf16
    const float* __restrict__ latent_bias,        // [F]
    unsigned short* __restrict__ lat_out,         // [B,F] bf16 (latents_pre)
    int* __restrict__ out_tidx,                   // [B,64]
    float* __restrict__ out_tval)                 // [B,64]
{
    const __bf16* Xd = reinterpret_cast<const __bf16*>(xd_in);
    const __bf16* WT = reinterpret_cast<const __bf16*>(wt_in);
    __bf16* lat = reinterpret_cast<__bf16*>(lat_out);

    __shared__ __bf16 Astrip[TM * DDIM];          // 48 KB
    __shared__ float  staging[TM * SSTRIDE];      // 16.5 KB
    __shared__ float  lval[TM][KTOP];             // 8 KB
    __shared__ int    lidx[TM][KTOP];             // 8 KB

    const int tid  = threadIdx.x;
    const int lane = tid & 31;
    const int wave = tid >> 5;                    // 8 waves
    const int b0   = blockIdx.x * TM;

    // ---- async copy of the 32-row x_diff strip: global -> LDS, no VGPR
    //      round trip, tracked with ASYNCcnt. 48 KB = 12 rounds x 256 x 16B.
    {
        const unsigned lds_base = (unsigned)(size_t)(&Astrip[0]);
        const char* gbase = (const char*)(Xd + (size_t)b0 * DDIM);
        #pragma unroll
        for (int it = 0; it < (TM * DDIM * 2) / (256 * 16); ++it) {
            unsigned loff = lds_base + (unsigned)((tid + it * 256) * 16);
            unsigned long long ga =
                (unsigned long long)(size_t)(gbase + (size_t)(tid + it * 256) * 16);
            asm volatile("global_load_async_to_lds_b128 %0, %1, off"
                         :: "v"(loff), "v"(ga) : "memory");
        }
    }

    // overlap: init top-k lists while the async copy is in flight
    for (int i = tid; i < TM * KTOP; i += 256) {
        (&lval[0][0])[i] = -3.0e38f;
        (&lidx[0][0])[i] = -1;
    }
    asm volatile("s_wait_asynccnt 0x0" ::: "memory");
    __syncthreads();

    // Each wave exclusively owns rows 4*wave .. 4*wave+3 for top-k.
    float rowMin[4]    = {-3.0e38f, -3.0e38f, -3.0e38f, -3.0e38f};
    int   rowMinPos[4] = {0, 0, 0, 0};

    const int khalf = (lane >= 16) ? 8 : 0;       // ISA 16-bit operand K split
    const __bf16* arow0 = Astrip + (lane & 15) * DDIM;         // rows 0..15
    const __bf16* arow1 = arow0 + 16 * DDIM;                   // rows 16..31

    for (int fc = 0; fc < FDIM; fc += NF) {
        // ---- GEMM: wave computes cols [fc + wave*16, +16) x 32 rows
        {
            const int n0  = fc + wave * 16;
            const int col = n0 + (lane & 15);
            const __bf16* bbase = WT + (size_t)col * DDIM;
            __builtin_prefetch(bbase, 0, 1);

            v8f acc0 = {0.f, 0.f, 0.f, 0.f, 0.f, 0.f, 0.f, 0.f};
            v8f acc1 = {0.f, 0.f, 0.f, 0.f, 0.f, 0.f, 0.f, 0.f};

            // software pipeline: B fragments for step k+32 issued before the
            // WMMAs of step k, hiding global-load latency behind matrix math.
            v8bf nb0 = *(const v8bf*)(bbase + khalf);
            v8bf nb1 = *(const v8bf*)(bbase + khalf + 16);
            #pragma unroll 2
            for (int k = 0; k < DDIM; k += 32) {
                v8bf cb0 = nb0, cb1 = nb1;
                const int kn = k + 32;
                if (kn < DDIM) {
                    nb0 = *(const v8bf*)(bbase + kn + khalf);
                    nb1 = *(const v8bf*)(bbase + kn + khalf + 16);
                }
                v16bf bv = __builtin_shufflevector(cb0, cb1,
                    0,1,2,3,4,5,6,7,8,9,10,11,12,13,14,15);

                v8bf a00 = *(const v8bf*)(arow0 + k + khalf);
                v8bf a01 = *(const v8bf*)(arow0 + k + khalf + 16);
                v16bf av0 = __builtin_shufflevector(a00, a01,
                    0,1,2,3,4,5,6,7,8,9,10,11,12,13,14,15);
                acc0 = __builtin_amdgcn_wmma_f32_16x16x32_bf16(
                    false, av0, false, bv, (short)0, acc0, false, false);

                v8bf a10 = *(const v8bf*)(arow1 + k + khalf);
                v8bf a11 = *(const v8bf*)(arow1 + k + khalf + 16);
                v16bf av1 = __builtin_shufflevector(a10, a11,
                    0,1,2,3,4,5,6,7,8,9,10,11,12,13,14,15);
                acc1 = __builtin_amdgcn_wmma_f32_16x16x32_bf16(
                    false, av1, false, bv, (short)0, acc1, false, false);
            }

            // C layout: lanes 0-15 -> M=v, N=lane; lanes 16-31 -> M=v+8
            const int mbase = (lane >= 16) ? 8 : 0;
            const int scol  = wave * 16 + (lane & 15);
            #pragma unroll
            for (int v = 0; v < 8; ++v) {
                staging[(mbase + v) * SSTRIDE + scol]        = acc0[v];
                staging[(16 + mbase + v) * SSTRIDE + scol]   = acc1[v];
            }
        }
        __syncthreads();

        // ---- spill latents_pre (no latent_bias) as bf16, coalesced
        for (int i = tid; i < TM * NF; i += 256) {
            int r = i >> 7, c = i & 127;
            lat[(size_t)(b0 + r) * FDIM + fc + c] = (__bf16)staging[r * SSTRIDE + c];
        }

        // ---- fused top-64 update (per-wave rows, replace-min list in LDS)
        #pragma unroll
        for (int rr = 0; rr < 4; ++rr) {
            const int r = wave * 4 + rr;
            for (int cbase = 0; cbase < NF; cbase += 32) {
                const int c = cbase + lane;
                const int f = fc + c;
                float vb = staging[r * SSTRIDE + c] + latent_bias[f];
                bool alive = true;
                while (true) {
                    unsigned long long m = __ballot(alive && (vb > rowMin[rr]));
                    if (m == 0ull) break;
                    int j = (int)__builtin_ctzll(m);
                    float cv = __shfl(vb, j);
                    int   ci = __shfl(f, j);
                    if (lane == j) alive = false;
                    if (lane == 0) { lval[r][rowMinPos[rr]] = cv;
                                     lidx[r][rowMinPos[rr]] = ci; }
                    // wave-parallel argmin rescan (2 entries / lane)
                    float e0 = lval[r][lane], e1 = lval[r][lane + 32];
                    float mv; int mp;
                    if (e0 <= e1) { mv = e0; mp = lane; }
                    else          { mv = e1; mp = lane + 32; }
                    #pragma unroll
                    for (int s = 16; s >= 1; s >>= 1) {
                        float ov = __shfl_xor(mv, s);
                        int   op = __shfl_xor(mp, s);
                        if (ov < mv) { mv = ov; mp = op; }
                    }
                    rowMin[rr] = mv; rowMinPos[rr] = mp;
                }
            }
        }
        __syncthreads();
    }

    // ---- sort descending + relu + emit
    #pragma unroll
    for (int rr = 0; rr < 4; ++rr) {
        const int r = wave * 4 + rr;
        for (int o = 0; o < KTOP; ++o) {
            float e0 = lval[r][lane], e1 = lval[r][lane + 32];
            float mv; int mp;
            if (e0 >= e1) { mv = e0; mp = lane; }
            else          { mv = e1; mp = lane + 32; }
            #pragma unroll
            for (int s = 16; s >= 1; s >>= 1) {
                float ov = __shfl_xor(mv, s);
                int   op = __shfl_xor(mp, s);
                if (ov > mv || (ov == mv && op < mp)) { mv = ov; mp = op; }
            }
            if (lane == 0) {
                out_tval[(size_t)(b0 + r) * KTOP + o] = mv > 0.f ? mv : 0.f;
                out_tidx[(size_t)(b0 + r) * KTOP + o] = lidx[r][mp];
                lval[r][mp] = -3.0e38f;
            }
        }
    }
}

// ------------------------------------------------------------------ stats ---
__global__ __launch_bounds__(256) void stats_init_kernel(
    const int* __restrict__ old_stats, int* __restrict__ out_stats)
{
    int f = blockIdx.x * 256 + threadIdx.x;
    if (f < FDIM) out_stats[f] = old_stats[f] + 1;
}

__global__ __launch_bounds__(256) void stats_hits_kernel(
    const float* __restrict__ tval, const int* __restrict__ tidx,
    int* __restrict__ out_stats)
{
    int i = blockIdx.x * 256 + threadIdx.x;
    if (i < BDIM * KTOP) {
        if (tval[i] > 0.001f) out_stats[tidx[i]] = 1;
    }
}

// ------------------------------------------------------------------- auxk ---
__global__ __launch_bounds__(256, 1) void auxk_kernel(
    const unsigned short* __restrict__ lat_in,    // [B,F] bf16
    const int* __restrict__ stats,                // stats_new
    int* __restrict__ out_aidx, float* __restrict__ out_aval,
    int dead_thresh)
{
    const __bf16* lat = reinterpret_cast<const __bf16*>(lat_in);
    __shared__ float cval[8 * AUXK];
    __shared__ int   cidx[8 * AUXK];
    __shared__ float redv[8];
    __shared__ int   redp[8];

    const int tid = threadIdx.x, lane = tid & 31, wave = tid >> 5;
    const int b = blockIdx.x;

    for (int i = lane; i < AUXK; i += 32) {
        cval[wave * AUXK + i] = -3.0e38f;
        cidx[wave * AUXK + i] = -1;
    }
    float mn = -3.0e38f; int mnp = 0;

    // per-wave top-256 over an F/8 slice
    const int fbase = wave * (FDIM / 8);
    for (int c = lane; c < FDIM / 8; c += 32) {
        int f = fbase + c;
        float lv = (float)lat[(size_t)b * FDIM + f];
        float v  = (stats[f] > dead_thresh) ? lv : 0.0f;
        bool alive = true;
        while (true) {
            unsigned long long m = __ballot(alive && (v > mn));
            if (m == 0ull) break;
            int j = (int)__builtin_ctzll(m);
            float cv2 = __shfl(v, j);
            int   ci  = __shfl(f, j);
            if (lane == j) alive = false;
            if (lane == 0) { cval[wave * AUXK + mnp] = cv2;
                             cidx[wave * AUXK + mnp] = ci; }
            float bmv = 3.0e38f; int bmp = 0;
            #pragma unroll
            for (int t = 0; t < 8; ++t) {
                int p = lane + t * 32;
                float e = cval[wave * AUXK + p];
                if (e < bmv) { bmv = e; bmp = p; }
            }
            #pragma unroll
            for (int s = 16; s >= 1; s >>= 1) {
                float ov = __shfl_xor(bmv, s);
                int   op = __shfl_xor(bmp, s);
                if (ov < bmv) { bmv = ov; bmp = op; }
            }
            mn = bmv; mnp = bmp;
        }
    }
    __syncthreads();

    // merge 8x256 candidates -> sorted top-256 (iterative extraction)
    for (int o = 0; o < AUXK; ++o) {
        float bmv = -3.0e38f; int bmp = wave * AUXK;
        #pragma unroll
        for (int t = 0; t < 8; ++t) {
            int p = wave * AUXK + lane + t * 32;
            float e = cval[p];
            if (e > bmv) { bmv = e; bmp = p; }
        }
        #pragma unroll
        for (int s = 16; s >= 1; s >>= 1) {
            float ov = __shfl_xor(bmv, s);
            int   op = __shfl_xor(bmp, s);
            if (ov > bmv) { bmv = ov; bmp = op; }
        }
        if (lane == 0) { redv[wave] = bmv; redp[wave] = bmp; }
        __syncthreads();
        if (tid == 0) {
            float best = redv[0]; int bp = redp[0];
            #pragma unroll
            for (int w = 1; w < 8; ++w)
                if (redv[w] > best) { best = redv[w]; bp = redp[w]; }
            out_aval[(size_t)b * AUXK + o] = best > 0.f ? best : 0.f;
            out_aidx[(size_t)b * AUXK + o] = cidx[bp];
            cval[bp] = -3.0e38f;
        }
        __syncthreads();
    }
}

// ----------------------------------------------------------------- decode ---
__global__ __launch_bounds__(256, 1) void decode_kernel(
    const float* __restrict__ W_dec, const float* __restrict__ pre_bias,
    const int* __restrict__ tidx, const float* __restrict__ tval,
    float* __restrict__ recons)
{
    __shared__ float svals[KTOP];
    __shared__ int   sidx[KTOP];
    const int b = blockIdx.x, tid = threadIdx.x;
    if (tid < KTOP) {
        svals[tid] = tval[(size_t)b * KTOP + tid];
        sidx[tid]  = tidx[(size_t)b * KTOP + tid];
    }
    __syncthreads();
    float a0 = 0.f, a1 = 0.f, a2 = 0.f;
    const int c0 = tid, c1 = tid + 256, c2 = tid + 512;
    for (int j = 0; j < KTOP; ++j) {
        float v = svals[j];
        const float* row = W_dec + (size_t)sidx[j] * DDIM;
        if (j + 1 < KTOP)
            __builtin_prefetch(W_dec + (size_t)sidx[j + 1] * DDIM + c0, 0, 0);
        a0 += v * row[c0];
        a1 += v * row[c1];
        a2 += v * row[c2];
    }
    recons[(size_t)b * DDIM + c0] = a0 + pre_bias[c0];
    recons[(size_t)b * DDIM + c1] = a1 + pre_bias[c1];
    recons[(size_t)b * DDIM + c2] = a2 + pre_bias[c2];
}

// ---------------------------------------------------------------- launch ----
extern "C" void kernel_launch(void* const* d_in, const int* in_sizes, int n_in,
                              void* d_out, int out_size, void* d_ws, size_t ws_size,
                              hipStream_t stream)
{
    (void)in_sizes; (void)n_in; (void)out_size; (void)ws_size;

    const float* x          = (const float*)d_in[0];   // [B,D]
    const float* pre_bias   = (const float*)d_in[1];   // [D]
    const float* W_enc      = (const float*)d_in[2];   // [D,F]
    const float* lat_bias   = (const float*)d_in[3];   // [F]
    const float* W_dec      = (const float*)d_in[4];   // [F,D]
    const int*   stats_old  = (const int*)d_in[5];     // [F]
    // d_in[6]=k, d_in[7]=auxk, d_in[8]=dead_steps_threshold (scalars)

    // Output layout (concatenated, return order)
    float* out_recons = (float*)d_out;                              // B*D
    int*   out_tidx   = (int*)(out_recons + (size_t)BDIM * DDIM);   // B*64
    float* out_tval   = (float*)(out_tidx + (size_t)BDIM * KTOP);   // B*64
    int*   out_aidx   = (int*)(out_tval + (size_t)BDIM * KTOP);     // B*256
    float* out_aval   = (float*)(out_aidx + (size_t)BDIM * AUXK);   // B*256
    int*   out_stats  = (int*)(out_aval + (size_t)BDIM * AUXK);     // F

    // Workspace layout
    char* ws = (char*)d_ws;
    unsigned short* xd   = (unsigned short*)(ws);                            // 6 MB
    unsigned short* wT   = (unsigned short*)(ws + (size_t)8   * 1024 * 1024); // 96 MB
    unsigned short* latw = (unsigned short*)(ws + (size_t)112 * 1024 * 1024); // 512 MB

    // 1) x_diff -> bf16
    prep_x_kernel<<<(BDIM * DDIM + 255) / 256, 256, 0, stream>>>(x, pre_bias, xd);
    // 2) W_enc transpose -> bf16 [F,D]
    transpose_wenc_kernel<<<dim3(FDIM / 64, DDIM / 64), 256, 0, stream>>>(W_enc, wT);
    // 3) WMMA encode + fused top-64 + latent spill
    encode_topk_kernel<<<BDIM / TM, 256, 0, stream>>>(xd, wT, lat_bias, latw,
                                                      out_tidx, out_tval);
    // 4) stats_new = old + 1
    stats_init_kernel<<<FDIM / 256, 256, 0, stream>>>(stats_old, out_stats);
    // 5) stats_new[idx]=1 on hits
    stats_hits_kernel<<<(BDIM * KTOP) / 256, 256, 0, stream>>>(out_tval, out_tidx,
                                                               out_stats);
    // 6) auxk top-256 over dead-masked latents
    auxk_kernel<<<BDIM, 256, 0, stream>>>(latw, out_stats, out_aidx, out_aval, 256);
    // 7) decode gather
    decode_kernel<<<BDIM, 256, 0, stream>>>(W_dec, pre_bias, out_tidx, out_tval,
                                            out_recons);
}
